// WishfulThinking_33200097198307
// MI455X (gfx1250) — matrix operand order repaired and checked by reading, not compiled, not run on hardware
//
#include <hip/hip_runtime.h>
#include <hip/hip_bf16.h>
#include <math.h>

#define N_NODES  20000
#define N_GRAPHS 32
#define E_LIG    100000
#define E_MAIN   320000

typedef __attribute__((ext_vector_type(16))) _Float16 v16h;
typedef __attribute__((ext_vector_type(8)))  float    v8f;

// ---------------------------------------------------------------- utilities
__global__ void fill_kernel(float* __restrict__ p, long n, float v) {
  long i = (long)blockIdx.x * blockDim.x + threadIdx.x;
  if (i < n) p[i] = v;
}

__device__ __forceinline__ void atomicMaxF(float* addr, float v) {
  if (v >= 0.f) atomicMax((int*)addr, __float_as_int(v));
  else          atomicMin((unsigned int*)addr, __float_as_uint(v));
}

// ------------------------------------------------- self-loop mean edge_attr
__global__ void deg_attr_kernel(const int* __restrict__ dst, const float* __restrict__ ea,
                                float* __restrict__ deg, float* __restrict__ msum, int E) {
  long i = (long)blockIdx.x * blockDim.x + threadIdx.x;
  if (i >= (long)E * 16) return;
  int e = (int)(i >> 4), k = (int)(i & 15);
  int d = dst[e];
  if (k == 0) atomicAdd(&deg[d], 1.f);
  atomicAdd(&msum[(long)d * 16 + k], ea[i]);
}

__global__ void mean_fin_kernel(float* __restrict__ msum, const float* __restrict__ deg, int N) {
  int i = blockIdx.x * blockDim.x + threadIdx.x;
  if (i >= N * 16) return;
  float dg = deg[i >> 4];
  msum[i] /= (dg > 1.f ? dg : 1.f);
}

// ------------------------------------------------- W -> f16 B-fragment pack
// Wp fragment f = kTile*(N/16)+nTile holds 32 lanes x 16 halves, contiguous:
//   Wp[(f*32+lane)*16 + h] = (f16) W[(kTile*32 + (h>>3)*16 + (lane>>4)*8 + (h&7)) * N + nTile*16 + (lane&15)]
__global__ void pack_w_kernel(const float* __restrict__ W, _Float16* __restrict__ Wp,
                              int K, int N) {
  long i = (long)blockIdx.x * blockDim.x + threadIdx.x;
  if (i >= (long)K * N) return;
  int h    = (int)(i & 15);
  int lane = (int)((i >> 4) & 31);
  long f   = i >> 9;
  int nTiles = N >> 4;
  int kTile = (int)(f / nTiles);
  int nTile = (int)(f % nTiles);
  int k = kTile * 32 + ((h >> 3) * 16) + ((lane >> 4) * 8) + (h & 7);
  int n = nTile * 16 + (lane & 15);
  Wp[i] = (_Float16)W[(long)k * N + n];
}

// ------------------------------------------------------------- WMMA GEMM
// C[M,N] = A[M,K] @ W[K,N] + bias[N].   Requires M%16==0, K%32==0, N%16==0.
// One wave per 16x16 tile; A from global as 4x float4/lane, B as one packed
// 32B fragment/lane; f32 accumulation via v_wmma_f32_16x16x32_f16.
__global__ void __launch_bounds__(32)
gemm_wmma_kernel(const float* __restrict__ A, const _Float16* __restrict__ Wp,
                 const float* __restrict__ bias, float* __restrict__ C,
                 int M, int K, int N) {
  const int lane   = threadIdx.x;
  const int mBase  = blockIdx.x * 16;
  const int nTile  = blockIdx.y;
  const int nTiles = N >> 4;

  const int aM   = mBase + (lane & 15);
  const int aOff = (lane >> 4) * 8;
  const float* arow = A + (long)aM * K;
  const v16h*  bp   = (const v16h*)Wp;

  v8f acc = {};
  for (int k0 = 0; k0 < K; k0 += 32) {
    const float4 a0 = *(const float4*)(arow + k0 + aOff);
    const float4 a1 = *(const float4*)(arow + k0 + aOff + 4);
    const float4 a2 = *(const float4*)(arow + k0 + 16 + aOff);
    const float4 a3 = *(const float4*)(arow + k0 + 16 + aOff + 4);
    const v16h b = bp[((long)(k0 >> 5) * nTiles + nTile) * 32 + lane];
    v16h a;
    a[0]  = (_Float16)a0.x; a[1]  = (_Float16)a0.y;
    a[2]  = (_Float16)a0.z; a[3]  = (_Float16)a0.w;
    a[4]  = (_Float16)a1.x; a[5]  = (_Float16)a1.y;
    a[6]  = (_Float16)a1.z; a[7]  = (_Float16)a1.w;
    a[8]  = (_Float16)a2.x; a[9]  = (_Float16)a2.y;
    a[10] = (_Float16)a2.z; a[11] = (_Float16)a2.w;
    a[12] = (_Float16)a3.x; a[13] = (_Float16)a3.y;
    a[14] = (_Float16)a3.z; a[15] = (_Float16)a3.w;
    acc = __builtin_amdgcn_wmma_f32_16x16x32_f16(false, a, false, b,
                                                 (short)0, acc, false, false);
  }

  const int outN   = nTile * 16 + (lane & 15);
  const int rowOff = (lane >> 4) * 8;                 // C/D layout: vgpr r -> M=r or r+8
  const float bv   = bias[outN];
#pragma unroll
  for (int r = 0; r < 8; ++r) {
    C[(long)(mBase + rowOff + r) * N + outN] = acc[r] + bv;
  }
}

// --------------------------------------------- per-edge attention logits
// ee = edge_attr @ We fused (rank-16 update) -> avoids 655 MB intermediate.
__global__ void __launch_bounds__(256)
edge_logits_kernel(const float* __restrict__ xl, const float* __restrict__ xr,
                   const float* __restrict__ ea, const float* __restrict__ ea_self,
                   const float* __restrict__ We, const float* __restrict__ att,
                   const int* __restrict__ src, const int* __restrict__ dst,
                   float* __restrict__ logit, int E_real, int E_tot,
                   int OUT, int H, int C) {
  const int lane = threadIdx.x & 31;
  const int e = blockIdx.x * (blockDim.x >> 5) + (threadIdx.x >> 5);
  if (e >= E_tot) return;

  int s, d; const float* earow;
  if (e < E_real) { s = src[e]; d = dst[e]; earow = ea + (long)e * 16; }
  else            { s = d = e - E_real;     earow = ea_self + (long)s * 16; }

  float eav[16];
#pragma unroll
  for (int k = 0; k < 16; ++k) eav[k] = earow[k];

  const float* xls = xl + (long)s * OUT;
  const float* xrd = xr + (long)d * OUT;

  for (int h = 0; h < H; ++h) {
    float part = 0.f;
    for (int c = lane; c < C; c += 32) {
      int ch = h * C + c;
      float eev = 0.f;
#pragma unroll
      for (int k = 0; k < 16; ++k) eev += eav[k] * We[k * OUT + ch];
      float m = xls[ch] + xrd[ch] + eev;
      m = (m > 0.f) ? m : 0.2f * m;                   // leaky_relu(0.2)
      part += att[h * C + c] * m;
    }
#pragma unroll
    for (int off = 16; off > 0; off >>= 1) part += __shfl_down(part, off, 32);
    if (lane == 0) logit[(long)e * H + h] = part;
  }
}

// ----------------------------------------------------- segment softmax
__global__ void seg_max_kernel(const float* __restrict__ logit, const int* __restrict__ dst,
                               float* __restrict__ mx, int E_real, int E_tot, int H) {
  long i = (long)blockIdx.x * blockDim.x + threadIdx.x;
  if (i >= (long)E_tot * H) return;
  int e = (int)(i / H), h = (int)(i % H);
  int d = (e < E_real) ? dst[e] : e - E_real;
  atomicMaxF(&mx[(long)d * H + h], logit[i]);
}

__global__ void seg_expsum_kernel(const float* __restrict__ logit, const int* __restrict__ dst,
                                  const float* __restrict__ mx, float* __restrict__ ex,
                                  float* __restrict__ den, int E_real, int E_tot, int H) {
  long i = (long)blockIdx.x * blockDim.x + threadIdx.x;
  if (i >= (long)E_tot * H) return;
  int e = (int)(i / H), h = (int)(i % H);
  int d = (e < E_real) ? dst[e] : e - E_real;
  float v = expf(logit[i] - mx[(long)d * H + h]);
  ex[i] = v;
  atomicAdd(&den[(long)d * H + h], v);
}

// --------------------------------------------------- alpha * xl scatter
__global__ void __launch_bounds__(256)
scatter_kernel(const float* __restrict__ ex, const float* __restrict__ den,
               const float* __restrict__ xl, const int* __restrict__ src,
               const int* __restrict__ dst, float* __restrict__ out,
               int E_real, int E_tot, int OUT, int H, int C) {
  const int lane = threadIdx.x & 31;
  const int e = blockIdx.x * (blockDim.x >> 5) + (threadIdx.x >> 5);
  if (e >= E_tot) return;
  int s, d;
  if (e < E_real) { s = src[e]; d = dst[e]; } else { s = d = e - E_real; }
  const float* xls = xl + (long)s * OUT;
  float* outd = out + (long)d * OUT;
  for (int h = 0; h < H; ++h) {
    float alpha = ex[(long)e * H + h] / den[(long)d * H + h];
    for (int c = lane; c < C; c += 32) {
      int ch = h * C + c;
      atomicAdd(&outd[ch], alpha * xls[ch]);
    }
  }
}

__global__ void bias_relu_kernel(float* __restrict__ p, const float* __restrict__ b,
                                 int N, int OUT) {
  long i = (long)blockIdx.x * blockDim.x + threadIdx.x;
  if (i >= (long)N * OUT) return;
  float v = p[i] + b[i % OUT];
  p[i] = (v > 0.f) ? v : 0.f;
}

// ------------------------------------------------------- pooling + head
__global__ void pool_kernel(const float* __restrict__ h, const int* __restrict__ batch,
                            float* __restrict__ pooled, int N) {
  long i = (long)blockIdx.x * blockDim.x + threadIdx.x;
  if (i >= (long)N * 256) return;
  int n = (int)(i >> 8), c = (int)(i & 255);
  atomicAdd(&pooled[(long)batch[n] * 256 + c], h[i]);
}

__global__ void __launch_bounds__(256)
head_kernel(const float* __restrict__ pooled, const float* __restrict__ w1,
            const float* __restrict__ b1, const float* __restrict__ w2,
            const float* __restrict__ b2, float* __restrict__ out) {
  __shared__ float g1[N_GRAPHS * 64];
  for (int i = threadIdx.x; i < N_GRAPHS * 64; i += blockDim.x) {
    int b = i >> 6, o = i & 63;
    float s = b1[o];
    for (int k = 0; k < 256; ++k) s += pooled[b * 256 + k] * w1[k * 64 + o];
    g1[i] = (s > 0.f) ? s : 0.f;
  }
  __syncthreads();
  for (int b = threadIdx.x; b < N_GRAPHS; b += blockDim.x) {
    float s = b2[0];
    for (int k = 0; k < 64; ++k) s += g1[b * 64 + k] * w2[k];
    out[b] = s;
  }
}

// ---------------------------------------------------------------- driver
static inline void fill(float* p, long n, float v, hipStream_t st) {
  fill_kernel<<<(unsigned)((n + 255) / 256), 256, 0, st>>>(p, n, v);
}

struct LayerW { const float *Wl, *bl, *Wr, *br, *We, *att, *bo; };

static void run_gat_layer(const float* xin, int IN, int OUT, int H, int C,
                          const int* src, const int* dst, const float* ea, int E_real,
                          const LayerW& w, float* XL, float* XR, float* DEG, float* MSUM,
                          float* LOGIT, float* EX, float* MX, float* DEN,
                          _Float16* WPL, _Float16* WPR, float* outb, hipStream_t st) {
  const int N = N_NODES;
  const int E_tot = E_real + N;

  // self-loop mean edge_attr
  fill(DEG, N, 0.f, st);
  fill(MSUM, (long)N * 16, 0.f, st);
  deg_attr_kernel<<<(unsigned)(((long)E_real * 16 + 255) / 256), 256, 0, st>>>(dst, ea, DEG, MSUM, E_real);
  mean_fin_kernel<<<(N * 16 + 255) / 256, 256, 0, st>>>(MSUM, DEG, N);

  // pack weights to f16 WMMA B-fragments, then xl = x@Wl+bl, xr = x@Wr+br
  long kn = (long)IN * OUT;
  pack_w_kernel<<<(unsigned)((kn + 255) / 256), 256, 0, st>>>(w.Wl, WPL, IN, OUT);
  pack_w_kernel<<<(unsigned)((kn + 255) / 256), 256, 0, st>>>(w.Wr, WPR, IN, OUT);
  dim3 gg(N / 16, OUT / 16);
  gemm_wmma_kernel<<<gg, 32, 0, st>>>(xin, WPL, w.bl, XL, N, IN, OUT);
  gemm_wmma_kernel<<<gg, 32, 0, st>>>(xin, WPR, w.br, XR, N, IN, OUT);

  // attention logits (edge transform fused)
  const int wpb = 8;  // waves per 256-thread block
  edge_logits_kernel<<<(E_tot + wpb - 1) / wpb, wpb * 32, 0, st>>>(
      XL, XR, ea, MSUM, w.We, w.att, src, dst, LOGIT, E_real, E_tot, OUT, H, C);

  // segment softmax
  fill(MX, (long)N * H, -INFINITY, st);
  fill(DEN, (long)N * H, 0.f, st);
  long nh = (long)E_tot * H;
  seg_max_kernel<<<(unsigned)((nh + 255) / 256), 256, 0, st>>>(LOGIT, dst, MX, E_real, E_tot, H);
  seg_expsum_kernel<<<(unsigned)((nh + 255) / 256), 256, 0, st>>>(LOGIT, dst, MX, EX, DEN, E_real, E_tot, H);

  // weighted scatter + bias + relu
  fill(outb, (long)N * OUT, 0.f, st);
  scatter_kernel<<<(E_tot + wpb - 1) / wpb, wpb * 32, 0, st>>>(
      EX, DEN, XL, src, dst, outb, E_real, E_tot, OUT, H, C);
  bias_relu_kernel<<<(unsigned)(((long)N * OUT + 255) / 256), 256, 0, st>>>(outb, w.bo, N, OUT);
}

extern "C" void kernel_launch(void* const* d_in, const int* in_sizes, int n_in,
                              void* d_out, int out_size, void* d_ws, size_t ws_size,
                              hipStream_t stream) {
  const float* x        = (const float*)d_in[0];
  const int*   src_lig  = (const int*)d_in[1];
  const int*   dst_lig  = src_lig + E_LIG;
  const float* ea_lig   = (const float*)d_in[2];
  const int*   src_main = (const int*)d_in[3];
  const int*   dst_main = src_main + E_MAIN;
  const float* ea_main  = (const float*)d_in[4];
  const int*   batch    = (const int*)d_in[5];

  LayerW L1 = {(const float*)d_in[6],  (const float*)d_in[7],  (const float*)d_in[8],
               (const float*)d_in[9],  (const float*)d_in[10], (const float*)d_in[11],
               (const float*)d_in[12]};
  LayerW L2 = {(const float*)d_in[13], (const float*)d_in[14], (const float*)d_in[15],
               (const float*)d_in[16], (const float*)d_in[17], (const float*)d_in[18],
               (const float*)d_in[19]};
  LayerW L3 = {(const float*)d_in[20], (const float*)d_in[21], (const float*)d_in[22],
               (const float*)d_in[23], (const float*)d_in[24], (const float*)d_in[25],
               (const float*)d_in[26]};
  const float* fc1_w = (const float*)d_in[27];
  const float* fc1_b = (const float*)d_in[28];
  const float* fc2_w = (const float*)d_in[29];
  const float* fc2_b = (const float*)d_in[30];

  // workspace carve-out (float units, 256B aligned blocks)
  float* ws = (float*)d_ws;
  size_t off = 0;
  auto carve = [&](size_t n) { float* p = ws + off; off += (n + 63) & ~(size_t)63; return p; };
  float* XL    = carve((size_t)N_NODES * 512);
  float* XR    = carve((size_t)N_NODES * 512);
  float* HB1   = carve((size_t)N_NODES * 512);   // h1, later h3
  float* HB2   = carve((size_t)N_NODES * 512);   // h2
  float* DEG   = carve(N_NODES);
  float* MSUM  = carve((size_t)N_NODES * 16);
  float* LOGIT = carve((size_t)(E_MAIN + N_NODES) * 4);
  float* EX    = carve((size_t)(E_MAIN + N_NODES) * 4);
  float* MX    = carve((size_t)N_NODES * 4);
  float* DEN   = carve((size_t)N_NODES * 4);
  float* POOL  = carve((size_t)N_GRAPHS * 256);
  _Float16* WPL = (_Float16*)carve((size_t)512 * 512 / 2);   // packed f16 weights
  _Float16* WPR = (_Float16*)carve((size_t)512 * 512 / 2);

  // conv1: 256 -> 256, H=1, ligand edges
  run_gat_layer(x,   256, 256, 1, 256, src_lig,  dst_lig,  ea_lig,  E_LIG,
                L1, XL, XR, DEG, MSUM, LOGIT, EX, MX, DEN, WPL, WPR, HB1, stream);
  // conv2: 256 -> 512 (H=4, C=128), main edges
  run_gat_layer(HB1, 256, 512, 4, 128, src_main, dst_main, ea_main, E_MAIN,
                L2, XL, XR, DEG, MSUM, LOGIT, EX, MX, DEN, WPL, WPR, HB2, stream);
  // conv3: 512 -> 256 (H=4, C=64), main edges; HB1 reusable now
  run_gat_layer(HB2, 512, 256, 4, 64,  src_main, dst_main, ea_main, E_MAIN,
                L3, XL, XR, DEG, MSUM, LOGIT, EX, MX, DEN, WPL, WPR, HB1, stream);

  // global add pool + MLP head
  fill(POOL, (long)N_GRAPHS * 256, 0.f, stream);
  pool_kernel<<<(unsigned)(((long)N_NODES * 256 + 255) / 256), 256, 0, stream>>>(HB1, batch, POOL, N_NODES);
  head_kernel<<<1, 256, 0, stream>>>(POOL, fc1_w, fc1_b, fc2_w, fc2_b, (float*)d_out);
}
